// SelfAttention_67095979099222
// MI455X (gfx1250) — matrix operand order repaired and checked by reading, not compiled, not run on hardware
//
#include <hip/hip_runtime.h>
#include <hip/hip_bf16.h>

typedef _Float16 h8   __attribute__((ext_vector_type(8)));
typedef _Float16 v16h __attribute__((ext_vector_type(16)));
typedef float    v8f  __attribute__((ext_vector_type(8)));
typedef float    f4   __attribute__((ext_vector_type(4)));
typedef int      v4i  __attribute__((ext_vector_type(4)));
typedef v4i __attribute__((address_space(1))) gv4i;   // global
typedef v4i __attribute__((address_space(3))) sv4i;   // LDS

#define B_   8
#define C_   256
#define HID_ 64
#define N_   4096

// ---- gfx1250 async global->LDS path (confirmed present on this toolchain)
#if defined(__gfx1250__) && __has_builtin(__builtin_amdgcn_global_load_async_to_lds_b128)
#define HAVE_ASYNC 1
#else
#define HAVE_ASYNC 0
#endif

#if HAVE_ASYNC
#if __has_builtin(__builtin_amdgcn_s_wait_asynccnt)
#define ASYNC_WAIT(n) __builtin_amdgcn_s_wait_asynccnt(n)
#else
#define ASYNC_WAIT(n) asm volatile("s_wait_asynccnt 0" ::: "memory")
#endif
#else
#define ASYNC_WAIT(n) ((void)0)
#endif

static __device__ __forceinline__ void cp_b128(_Float16* lds_dst, const _Float16* gsrc) {
#if HAVE_ASYNC
    __builtin_amdgcn_global_load_async_to_lds_b128((gv4i*)gsrc, (sv4i*)lds_dst, 0, 0);
#else
    *(f4*)lds_dst = *(const f4*)gsrc;
#endif
}

static __device__ __forceinline__ v16h cat8(h8 lo, h8 hi) {
    v16h r;
#pragma unroll
    for (int i = 0; i < 8; ++i) { r[i] = lo[i]; r[i + 8] = hi[i]; }
    return r;
}

static __device__ __forceinline__ v8f wmma_f16(v16h a, v16h b, v8f c) {
    return __builtin_amdgcn_wmma_f32_16x16x32_f16(false, a, false, b, (short)0, c,
                                                  false, false);
}

// ---------------------------------------------------------------------------
// Prep: xh[b][n][c] = (f16) x[b][c][n] -- LDS-tiled 32x32 transpose.
// ---------------------------------------------------------------------------
__global__ void k_prep_x(const float* __restrict__ x, _Float16* __restrict__ xh) {
    __shared__ float tl[32][33];
    const int tid = threadIdx.x;                 // 256 threads
    const int tx  = tid & 31;
    const int ty  = tid >> 5;                    // 0..7
    const int b   = blockIdx.x >> 10;            // 8
    const int ct  = (blockIdx.x >> 7) & 7;       // 8 c-tiles
    const int nt  = blockIdx.x & 127;            // 128 n-tiles

#pragma unroll
    for (int i = 0; i < 4; ++i) {                // read along n (coalesced)
        int c = ct * 32 + ty * 4 + i;
        tl[ty * 4 + i][tx] = x[((size_t)b * C_ + c) * N_ + nt * 32 + tx];
    }
    __syncthreads();
#pragma unroll
    for (int i = 0; i < 4; ++i) {                // write along c (coalesced)
        int n = nt * 32 + ty * 4 + i;
        xh[((size_t)b * N_ + n) * C_ + ct * 32 + tx] = (_Float16)tl[tx][ty * 4 + i];
    }
}

// ---------------------------------------------------------------------------
// Prep: weights -> tap-major f16: Wt[tap][c_out][c_in]
// ---------------------------------------------------------------------------
__global__ void k_prep_w(const float* __restrict__ Wq, const float* __restrict__ Wk,
                         const float* __restrict__ Wv,
                         _Float16* __restrict__ WqT, _Float16* __restrict__ WkT,
                         _Float16* __restrict__ WvT) {
    int idx = blockIdx.x * 256 + threadIdx.x;    // 2*147456 + 589824 = 884736
    if (idx < 2 * 147456) {
        const float* W  = (idx < 147456) ? Wq : Wk;
        _Float16*    Wt = (idx < 147456) ? WqT : WkT;
        int i   = (idx < 147456) ? idx : idx - 147456;
        int ci  = i & 255;
        int co  = (i >> 8) & 63;
        int tap = i >> 14;
        Wt[i] = (_Float16)W[(co * C_ + ci) * 9 + tap];
    } else {
        int i   = idx - 2 * 147456;
        int ci  = i & 255;
        int co  = (i >> 8) & 255;
        int tap = i >> 16;
        WvT[i] = (_Float16)Wv[(co * C_ + ci) * 9 + tap];
    }
}

// ---------------------------------------------------------------------------
// Conv q,k: implicit GEMM, weights async-staged to LDS per (tap, k-half).
// ---------------------------------------------------------------------------
__global__ void __launch_bounds__(128, 1)
k_conv_qk(const _Float16* __restrict__ xh,
          const _Float16* __restrict__ WqT, const float* __restrict__ bq,
          const _Float16* __restrict__ WkT, const float* __restrict__ bk,
          _Float16* __restrict__ Qh, _Float16* __restrict__ KTm) {
    __shared__ __align__(32) _Float16 Wl[2][2][64 * 128];   // 64KB
    const int tid  = threadIdx.x;
    const int lane = tid & 31;
    const int wave = tid >> 5;
    const int tile = blockIdx.x * 4 + wave;
    const int b    = tile >> 8;
    const int m0   = (tile & 255) * 16;
    const int col  = lane & 15;
    const int half = lane >> 4;
    const int y    = m0 >> 6;
    const int xx   = (m0 & 63) + col;
    const _Float16* xb = xh + (size_t)b * N_ * C_;

    auto issue_step = [&](int buf, int st) {
        int tap = st >> 1, kh = st & 1;
        const _Float16* sq = WqT + tap * (HID_ * C_) + kh * 128;
        const _Float16* sk = WkT + tap * (HID_ * C_) + kh * 128;
#pragma unroll
        for (int s = 0; s < 8; ++s) {
            int q   = tid + s * 128;
            int row = q >> 4;
            int off = (q & 15) * 8;
            cp_b128(&Wl[buf][0][q * 8], sq + row * C_ + off);
            cp_b128(&Wl[buf][1][q * 8], sk + row * C_ + off);
        }
    };

    v8f  zero = {};
    v16h azero = {};
    v8f  aq[4], ak[4];
#pragma unroll
    for (int t = 0; t < 4; ++t) { aq[t] = zero; ak[t] = zero; }

    issue_step(0, 0);
    for (int st = 0; st < 18; ++st) {
        if (st + 1 < 18) { issue_step((st + 1) & 1, st + 1); ASYNC_WAIT(16); }
        else             { ASYNC_WAIT(0); }
        __syncthreads();

        int  tap = st >> 1, kh = st & 1;
        int  dy = tap / 3 - 1, dx = tap % 3 - 1;
        int  y2 = y + dy, x2 = xx + dx;
        bool valid = ((unsigned)y2 < 64u) & ((unsigned)x2 < 64u);
        const _Float16* arow = xb + (y2 * 64 + x2) * C_ + kh * 128;
        const _Float16* wq   = &Wl[st & 1][0][col * 128];
        const _Float16* wk   = &Wl[st & 1][1][col * 128];

        v16h a = azero;
        if (valid) a = cat8(*(const h8*)(arow + half * 16),
                            *(const h8*)(arow + 8 + half * 16));
        for (int kl = 0; kl < 128; kl += 32) {
            v16h bt[4];                          // batch of 4 B tiles (q)
#pragma unroll
            for (int t = 0; t < 4; ++t)
                bt[t] = *(const v16h*)(wq + t * 16 * 128 + kl + half * 16);
            v16h an = azero;                     // prefetch next A k-step
            if (valid && (kl + 32 < 128))
                an = cat8(*(const h8*)(arow + kl + 32 + half * 16),
                          *(const h8*)(arow + kl + 40 + half * 16));
#pragma unroll
            for (int t = 0; t < 4; ++t)
                aq[t] = wmma_f16(a, bt[t], aq[t]);
#pragma unroll
            for (int t = 0; t < 4; ++t)          // batch of 4 B tiles (k)
                bt[t] = *(const v16h*)(wk + t * 16 * 128 + kl + half * 16);
#pragma unroll
            for (int t = 0; t < 4; ++t)
                ak[t] = wmma_f16(a, bt[t], ak[t]);
            a = an;
        }
        __syncthreads();
    }

    _Float16* qb = Qh  + (size_t)b * N_ * HID_;
    _Float16* kb = KTm + (size_t)b * N_ * HID_;
#pragma unroll
    for (int t = 0; t < 4; ++t) {
        int   c   = t * 16 + col;
        float bqv = bq[c], bkv = bk[c];
#pragma unroll
        for (int r = 0; r < 8; ++r) {
            int m = m0 + r + 8 * half;
            qb[m * HID_ + c] = (_Float16)(aq[t][r] + bqv);
            kb[m * HID_ + c] = (_Float16)(ak[t][r] + bkv);
        }
    }
}

// ---------------------------------------------------------------------------
// Conv v: 4 waves share one weight block; output [C][N] with packed h8 stores.
// ---------------------------------------------------------------------------
__global__ void __launch_bounds__(128, 1)
k_conv_v(const _Float16* __restrict__ xh,
         const _Float16* __restrict__ WvT, const float* __restrict__ bv,
         _Float16* __restrict__ Vh) {
    __shared__ __align__(32) _Float16 Wl[2][64 * 128];      // 32KB
    const int tid  = threadIdx.x;
    const int lane = tid & 31;
    const int wave = tid >> 5;
    const int cb   = blockIdx.x & 3;
    const int g    = (blockIdx.x >> 2) & 63;
    const int b    = blockIdx.x >> 8;
    const int m0   = (g * 4 + wave) * 16;
    const int col  = lane & 15;
    const int half = lane >> 4;
    const int y    = m0 >> 6;
    const int xx   = (m0 & 63) + col;
    const _Float16* xb = xh + (size_t)b * N_ * C_;

    auto issue_step = [&](int buf, int st) {
        int tap = st >> 1, kh = st & 1;
        const _Float16* sv = WvT + tap * (C_ * C_) + (cb * 64) * C_ + kh * 128;
#pragma unroll
        for (int s = 0; s < 8; ++s) {
            int q   = tid + s * 128;
            int row = q >> 4;
            int off = (q & 15) * 8;
            cp_b128(&Wl[buf][q * 8], sv + row * C_ + off);
        }
    };

    v8f  zero = {};
    v16h azero = {};
    v8f  acc[4];
#pragma unroll
    for (int t = 0; t < 4; ++t) acc[t] = zero;

    issue_step(0, 0);
    for (int st = 0; st < 18; ++st) {
        if (st + 1 < 18) { issue_step((st + 1) & 1, st + 1); ASYNC_WAIT(8); }
        else             { ASYNC_WAIT(0); }
        __syncthreads();

        int  tap = st >> 1, kh = st & 1;
        int  dy = tap / 3 - 1, dx = tap % 3 - 1;
        int  y2 = y + dy, x2 = xx + dx;
        bool valid = ((unsigned)y2 < 64u) & ((unsigned)x2 < 64u);
        const _Float16* arow = xb + (y2 * 64 + x2) * C_ + kh * 128;
        const _Float16* wv   = &Wl[st & 1][col * 128];

        v16h a = azero;
        if (valid) a = cat8(*(const h8*)(arow + half * 16),
                            *(const h8*)(arow + 8 + half * 16));
        for (int kl = 0; kl < 128; kl += 32) {
            v16h bt[4];
#pragma unroll
            for (int t = 0; t < 4; ++t)
                bt[t] = *(const v16h*)(wv + t * 16 * 128 + kl + half * 16);
            v16h an = azero;
            if (valid && (kl + 32 < 128))
                an = cat8(*(const h8*)(arow + kl + 32 + half * 16),
                          *(const h8*)(arow + kl + 40 + half * 16));
#pragma unroll
            for (int t = 0; t < 4; ++t)
                acc[t] = wmma_f16(a, bt[t], acc[t]);
            a = an;
        }
        __syncthreads();
    }

    _Float16* vbp = Vh + (size_t)b * C_ * N_;
#pragma unroll
    for (int t = 0; t < 4; ++t) {
        int   c   = cb * 64 + t * 16 + col;
        float bvv = bv[c];
        h8    hv;
#pragma unroll
        for (int r = 0; r < 8; ++r) hv[r] = (_Float16)(acc[t][r] + bvv);
        *(h8*)(vbp + c * N_ + m0 + 8 * half) = hv;    // 8 contiguous m, one b128
    }
}

// ---------------------------------------------------------------------------
// Fused flash attention. K/V async-staged to LDS per block (double-buffered).
// ---------------------------------------------------------------------------
__global__ void __launch_bounds__(128, 1)
k_attn(const _Float16* __restrict__ Qh, const _Float16* __restrict__ KTm,
       const _Float16* __restrict__ Vh, const float* __restrict__ x,
       float* __restrict__ out) {
    __shared__ __align__(32) _Float16 Kl[2][32 * 64];    // 8 KB
    __shared__ __align__(32) _Float16 Vl[2][256 * 32];   // 32 KB
    __shared__ __align__(32) _Float16 plds[4][16 * 32];  // 4 KB
    const int tid  = threadIdx.x;
    const int lane = tid & 31;
    const int wave = tid >> 5;
    const int tile = blockIdx.x * 4 + wave;
    const int b    = tile >> 8;
    const int m0   = (tile & 255) * 16;
    const int col  = lane & 15;
    const int half = lane >> 4;

    const _Float16* ktb = KTm + (size_t)b * N_ * HID_;
    const _Float16* vb  = Vh  + (size_t)b * C_ * N_;

    auto issue_chunk = [&](int buf, int n0) {    // 4KB K + 16KB V
#pragma unroll
        for (int s = 0; s < 2; ++s) {
            int q = tid + s * 128;
            int row = q >> 3, off = (q & 7) * 8;
            cp_b128(&Kl[buf][q * 8], ktb + (n0 + row) * HID_ + off);
        }
#pragma unroll
        for (int s = 0; s < 8; ++s) {
            int q = tid + s * 128;
            int row = q >> 2, off = (q & 3) * 8;
            cp_b128(&Vl[buf][q * 8], vb + row * N_ + n0 + off);
        }
    };

    const _Float16* qrow = Qh + (size_t)b * N_ * HID_ + (m0 + col) * HID_;
    v16h aq0 = cat8(*(const h8*)(qrow + half * 16),      *(const h8*)(qrow + 8 + half * 16));
    v16h aq1 = cat8(*(const h8*)(qrow + 32 + half * 16), *(const h8*)(qrow + 40 + half * 16));

    v8f zero = {};
    v8f oacc[16];
#pragma unroll
    for (int t = 0; t < 16; ++t) oacc[t] = zero;
    float rmax[8], rsum[8];
#pragma unroll
    for (int r = 0; r < 8; ++r) { rmax[r] = -3.0e38f; rsum[r] = 0.0f; }

    issue_chunk(0, 0);
    for (int i = 0; i < 128; ++i) {
        int n0 = i * 32;
        if (i + 1 < 128) { issue_chunk((i + 1) & 1, n0 + 32); ASYNC_WAIT(10); }
        else             { ASYNC_WAIT(0); }
        __syncthreads();
        const _Float16* Kc = &Kl[i & 1][0];
        const _Float16* Vc = &Vl[i & 1][0];

        // ---- scores: batch all 4 K tiles, then 4 WMMAs
        v16h kb4[4];
        {
            const _Float16* k0r = Kc + col * HID_;
            const _Float16* k1r = Kc + (16 + col) * HID_;
            kb4[0] = *(const v16h*)(k0r + half * 16);
            kb4[1] = *(const v16h*)(k0r + 32 + half * 16);
            kb4[2] = *(const v16h*)(k1r + half * 16);
            kb4[3] = *(const v16h*)(k1r + 32 + half * 16);
        }
        v8f s0 = zero, s1 = zero;
        s0 = wmma_f16(aq0, kb4[0], s0);
        s0 = wmma_f16(aq1, kb4[1], s0);
        s1 = wmma_f16(aq0, kb4[2], s1);
        s1 = wmma_f16(aq1, kb4[3], s1);

        // ---- online softmax update
        float corr[8];
        int   anych = 0;
#pragma unroll
        for (int r = 0; r < 8; ++r) {
            float v = fmaxf(s0[r], s1[r]);
#pragma unroll
            for (int off = 1; off < 16; off <<= 1)
                v = fmaxf(v, __shfl_xor(v, off, 32));
            float nm = fmaxf(rmax[r], v);
            corr[r]  = __expf(rmax[r] - nm);
            anych   |= (corr[r] < 1.0f);
            rmax[r]  = nm;
            float p0 = __expf(s0[r] - nm);
            float p1 = __expf(s1[r] - nm);
            rsum[r]  = rsum[r] * corr[r] + p0 + p1;
            plds[wave][(r + 8 * half) * 32 + col]      = (_Float16)p0;
            plds[wave][(r + 8 * half) * 32 + 16 + col] = (_Float16)p1;
        }
        if (__any(anych)) {                      // rescale only when max moved
#pragma unroll
            for (int t = 0; t < 16; ++t)
#pragma unroll
                for (int r = 0; r < 8; ++r) oacc[t][r] *= corr[r];
        }

        // ---- P: C-layout -> A-layout through wave-private LDS tile
        asm volatile("s_wait_dscnt 0" ::: "memory");
        const _Float16* prow = &plds[wave][col * 32];
        v16h ap = cat8(*(const h8*)(prow + half * 16), *(const h8*)(prow + 8 + half * 16));

        // ---- O += P * V: four groups of (load 4 B tiles -> 4 WMMAs)
#pragma unroll
        for (int g = 0; g < 4; ++g) {
            v16h vbuf[4];
#pragma unroll
            for (int t = 0; t < 4; ++t)
                vbuf[t] = *(const v16h*)(Vc + ((g * 4 + t) * 16 + col) * 32 + half * 16);
#pragma unroll
            for (int t = 0; t < 4; ++t)
                oacc[g * 4 + t] = wmma_f16(ap, vbuf[t], oacc[g * 4 + t]);
        }
        __syncthreads();
    }

    // ---- final row sums, normalize, add residual, vectorized f32 store
    float inv[8];
#pragma unroll
    for (int r = 0; r < 8; ++r) {
        float v = rsum[r];
#pragma unroll
        for (int off = 1; off < 16; off <<= 1) v += __shfl_xor(v, off, 32);
        inv[r] = 1.0f / v;
    }
    const float* xb = x   + (size_t)b * C_ * N_;
    float*       ob = out + (size_t)b * C_ * N_;
    const int mbase = m0 + 8 * half;             // 8 contiguous m per lane
#pragma unroll
    for (int t = 0; t < 16; ++t) {
        int c = t * 16 + col;
        f4 x0 = *(const f4*)(xb + (size_t)c * N_ + mbase);
        f4 x1 = *(const f4*)(xb + (size_t)c * N_ + mbase + 4);
        f4 o0, o1;
#pragma unroll
        for (int j = 0; j < 4; ++j) {
            o0[j] = oacc[t][j]     * inv[j]     + x0[j];
            o1[j] = oacc[t][j + 4] * inv[j + 4] + x1[j];
        }
        *(f4*)(ob + (size_t)c * N_ + mbase)     = o0;
        *(f4*)(ob + (size_t)c * N_ + mbase + 4) = o1;
    }
}

// ---------------------------------------------------------------------------
extern "C" void kernel_launch(void* const* d_in, const int* in_sizes, int n_in,
                              void* d_out, int out_size, void* d_ws, size_t ws_size,
                              hipStream_t stream) {
    const float* x  = (const float*)d_in[0];
    const float* Wq = (const float*)d_in[1];
    const float* bq = (const float*)d_in[2];
    const float* Wk = (const float*)d_in[3];
    const float* bk = (const float*)d_in[4];
    const float* Wv = (const float*)d_in[5];
    const float* bv = (const float*)d_in[6];
    float* out = (float*)d_out;

    char*  ws  = (char*)d_ws;
    size_t off = 0;
    auto alloc = [&](size_t bytes) -> void* {
        void* p = ws + off;
        off += (bytes + 255) & ~size_t(255);
        return p;
    };
    _Float16* xh  = (_Float16*)alloc((size_t)B_ * N_ * C_ * 2);
    _Float16* WqT = (_Float16*)alloc((size_t)9 * HID_ * C_ * 2);
    _Float16* WkT = (_Float16*)alloc((size_t)9 * HID_ * C_ * 2);
    _Float16* WvT = (_Float16*)alloc((size_t)9 * C_ * C_ * 2);
    _Float16* Qh  = (_Float16*)alloc((size_t)B_ * N_ * HID_ * 2);
    _Float16* KTm = (_Float16*)alloc((size_t)B_ * N_ * HID_ * 2);
    _Float16* Vh  = (_Float16*)alloc((size_t)B_ * C_ * N_ * 2);

    k_prep_x<<<8192, 256, 0, stream>>>(x, xh);
    k_prep_w<<<884736 / 256, 256, 0, stream>>>(Wq, Wk, Wv, WqT, WkT, WvT);
    k_conv_qk<<<512, 128, 0, stream>>>(xh, WqT, bq, WkT, bk, Qh, KTm);
    k_conv_v<<<2048, 128, 0, stream>>>(xh, WvT, bv, Vh);
    k_attn<<<512, 128, 0, stream>>>(Qh, KTm, Vh, x, out);
}
